// Decoder_40364102648369
// MI455X (gfx1250) — matrix operand order repaired and checked by reading, not compiled, not run on hardware
//
#include <hip/hip_runtime.h>
#include <math.h>

// ---------------------------------------------------------------------------
// LSTM decoder (L=12, B=64, T=128, E=H=512, V=1000) on gfx1250.
// Wavefront (diagonal) schedule: one kernel launch per diagonal d = l + t.
// bf16 WMMA (v_wmma_f32_16x16x32_bf16) with fp32 accumulation; fp32 cell state.
// ---------------------------------------------------------------------------

#define L_LAYERS 12
#define B_BATCH  64
#define T_STEPS  128
#define E_DIM    512
#define H_DIM    512
#define V_VOCAB  1000
#define NT_OUT   63          // ceil(1000/16)
#define BOS_ID   1

typedef __attribute__((ext_vector_type(16))) __bf16 v16bf;
typedef __attribute__((ext_vector_type(8)))  __bf16 v8bf;
typedef __attribute__((ext_vector_type(8)))  float  v8f;

// Workspace layout (bytes)
#define WS_WSW    0                         // 12*2048*1024 bf16 = 50331648
#define WS_WOUT   50331648                  // 63*16*512 bf16    = 1032192
#define WS_XEMB   51363840                  // 128*64*512 bf16   = 8388608
#define WS_HBUF   59752448                  // 2*12*64*512 bf16  = 1572864
#define WS_CBUF   61325312                  // 12*64*512 f32     = 1572864

__device__ __forceinline__ v8f wmma_bf16(v16bf a, v16bf b, v8f c) {
  return __builtin_amdgcn_wmma_f32_16x16x32_bf16(
      /*neg_a=*/false, a, /*neg_b=*/false, b,
      /*c_mod=*/(short)0, c, /*reuse_a=*/false, /*reuse_b=*/false);
}

__device__ __forceinline__ float sigf(float x) {
  return 1.0f / (1.0f + __expf(-x));
}

// Load a 16x32 bf16 A-fragment row slice for this lane (contiguous k order, same as the B swizzle):
// lane group g = lane/16, m = base_m + lane%16,
// vector elems 0..15 -> k = kb + g*16 + e
__device__ __forceinline__ v16bf load_afrag(const __bf16* row, int kb, int g) {
  const v8bf lo = *(const v8bf*)(row + kb + g * 16);
  const v8bf hi = *(const v8bf*)(row + kb + g * 16 + 8);
  return __builtin_shufflevector(lo, hi, 0, 1, 2, 3, 4, 5, 6, 7,
                                 8, 9, 10, 11, 12, 13, 14, 15);
}

// ---------------------------------------------------------------------------
// Setup kernel 1: swizzle weights to bf16 WMMA B-fragment order.
// LSTM frag f = ((l*32 + jt)*4 + gate)*32 + kc ; per-frag 512 bf16, lane-major:
//   elem s = lane*16 + e ; n = gate*512 + jt*16 + lane%16 ;
//   k = kc*32 + (lane/16)*16 + e ; B[k][n] = Wcat[n][k]
// ---------------------------------------------------------------------------
__global__ __launch_bounds__(256) void swizzle_weights(
    const float* __restrict__ W_ih, const float* __restrict__ W_hh,
    const float* __restrict__ W_out, __bf16* __restrict__ Wsw,
    __bf16* __restrict__ WoutSw) {
  const int f = blockIdx.x;
  const int tid = threadIdx.x;
  if (f < L_LAYERS * 32 * 4 * 32) {
    const int kc = f & 31;
    const int gate = (f >> 5) & 3;
    const int jt = (f >> 7) & 31;
    const int l = f >> 12;
    for (int p = 0; p < 2; ++p) {
      const int s = tid + p * 256;
      const int lane = s >> 4, e = s & 15;
      const int n = gate * 512 + jt * 16 + (lane & 15);
      const int k = kc * 32 + (lane >> 4) * 16 + e;
      float v;
      if (k < 512)
        v = W_ih[((size_t)l * 2048 + n) * 512 + k];
      else
        v = W_hh[((size_t)l * 2048 + n) * 512 + (k - 512)];
      Wsw[(size_t)f * 512 + s] = (__bf16)v;
    }
  } else {
    const int f2 = f - L_LAYERS * 32 * 4 * 32;  // nt*16 + kc
    const int kc = f2 & 15;
    const int nt = f2 >> 4;
    for (int p = 0; p < 2; ++p) {
      const int s = tid + p * 256;
      const int lane = s >> 4, e = s & 15;
      const int n = nt * 16 + (lane & 15);
      const int k = kc * 32 + (lane >> 4) * 16 + e;
      const float v = (n < V_VOCAB) ? W_out[(size_t)n * 512 + k] : 0.0f;
      WoutSw[(size_t)f2 * 512 + s] = (__bf16)v;
    }
  }
}

// ---------------------------------------------------------------------------
// Setup kernel 2: teacher-forcing token shift + embedding gather -> bf16.
// ---------------------------------------------------------------------------
__global__ __launch_bounds__(256) void embed_tokens(
    const int* __restrict__ tok, const float* __restrict__ table,
    __bf16* __restrict__ Xemb) {
  const int bid = blockIdx.x;
  const int t = bid >> 6;        // / B
  const int b = bid & 63;
  const int id = (t == 0) ? BOS_ID : tok[b * T_STEPS + (t - 1)];
  const float* row = table + (size_t)id * E_DIM;
  __bf16* dst = Xemb + ((size_t)t * B_BATCH + b) * E_DIM;
  for (int e = threadIdx.x; e < E_DIM; e += 256) dst[e] = (__bf16)row[e];
}

// ---------------------------------------------------------------------------
// Setup kernel 3: broadcast h0/c0 to all layers (both h parities).
// ---------------------------------------------------------------------------
__global__ __launch_bounds__(256) void init_state(
    const float* __restrict__ h0, const float* __restrict__ c0,
    __bf16* __restrict__ hbuf, float* __restrict__ cbuf) {
  const int idx = blockIdx.x * 256 + threadIdx.x;
  if (idx >= L_LAYERS * B_BATCH * H_DIM) return;
  const int j = idx & 511;
  const int b = (idx >> 9) & 63;
  const float hv = h0[b * H_DIM + j];
  hbuf[idx] = (__bf16)hv;
  hbuf[idx + L_LAYERS * B_BATCH * H_DIM] = (__bf16)hv;
  cbuf[idx] = c0[b * H_DIM + j];
}

// ---------------------------------------------------------------------------
// One wavefront diagonal d = l + t  (l=12 is the vocab projection "layer").
// Blocks 0..95: LSTM cells, 8 WGs per layer; blocks 96..103: logits WGs.
// Wave = 32 lanes (wave32). Each LSTM wave: M=32 x 16 j-cols x 4 gates.
// ---------------------------------------------------------------------------
__global__ __launch_bounds__(256) void diag_step(
    int d, const float* __restrict__ b_ih, const float* __restrict__ b_hh,
    const float* __restrict__ b_out, const __bf16* __restrict__ Wsw,
    const __bf16* __restrict__ WoutSw, const __bf16* __restrict__ Xemb,
    __bf16* __restrict__ hbuf, float* __restrict__ cbuf,
    float* __restrict__ out) {
  const int tid = threadIdx.x;
  const int wave = tid >> 5;
  const int lane = tid & 31;
  const int lm = lane & 15;   // m / n offset within tile
  const int g = lane >> 4;    // lane group for A layout
  const int parR = (d + 1) & 1;
  const int parW = d & 1;
  const size_t HSLAB = (size_t)L_LAYERS * B_BATCH * H_DIM;

  if (blockIdx.x < 96) {
    // ---------------- LSTM cell (l, t) ----------------
    const int l = blockIdx.x >> 3;
    const int wg = blockIdx.x & 7;
    const int t = d - l;
    if (t < 0 || t >= T_STEPS) return;

    const __bf16* inrows =
        (l == 0) ? (Xemb + (size_t)t * B_BATCH * E_DIM)
                 : (hbuf + parR * HSLAB + (size_t)(l - 1) * B_BATCH * H_DIM);
    const __bf16* hrows = hbuf + parR * HSLAB + (size_t)l * B_BATCH * H_DIM;

    const int jt = wg * 4 + (wave & 3);  // 16-wide j sub-tile, 0..31
    const int mh = wave >> 2;            // 0 or 1 -> M base 0 / 32
    const int m_base = mh * 32;

    v8f acc[4][2];
    const v8f zero = {0.f, 0.f, 0.f, 0.f, 0.f, 0.f, 0.f, 0.f};
    for (int q = 0; q < 4; ++q) { acc[q][0] = zero; acc[q][1] = zero; }

    const __bf16* wbase = Wsw + (size_t)((l * 32 + jt) * 4) * 32 * 512;

    for (int kc = 0; kc < 32; ++kc) {
      const __bf16* src = (kc < 16) ? inrows : hrows;
      const int kb = (kc & 15) * 32;
      v16bf a0 = load_afrag(src + (size_t)(m_base + lm) * 512, kb, g);
      v16bf a1 = load_afrag(src + (size_t)(m_base + 16 + lm) * 512, kb, g);
      for (int gate = 0; gate < 4; ++gate) {
        const v16bf bf =
            *(const v16bf*)(wbase + (size_t)(gate * 32 + kc) * 512 + lane * 16);
        acc[gate][0] = wmma_bf16(a0, bf, acc[gate][0]);
        acc[gate][1] = wmma_bf16(a1, bf, acc[gate][1]);
      }
    }

    // Fused LSTM elementwise (fp32).
    const int j = jt * 16 + lm;
    float bias[4];
    for (int gate = 0; gate < 4; ++gate) {
      const size_t bidx = (size_t)l * 2048 + gate * 512 + j;
      bias[gate] = b_ih[bidx] + b_hh[bidx];
    }
    __bf16* hdst = hbuf + parW * HSLAB + (size_t)l * B_BATCH * H_DIM;
    for (int mt2 = 0; mt2 < 2; ++mt2) {
      for (int r = 0; r < 8; ++r) {
        const int m = m_base + mt2 * 16 + ((lane < 16) ? r : (r + 8));
        const size_t cidx = ((size_t)l * B_BATCH + m) * H_DIM + j;
        const float c_old = cbuf[cidx];
        const float iv = acc[0][mt2][r] + bias[0];
        const float fv = acc[1][mt2][r] + bias[1];
        const float gv = acc[2][mt2][r] + bias[2];
        const float ov = acc[3][mt2][r] + bias[3];
        const float cn = sigf(fv) * c_old + sigf(iv) * tanhf(gv);
        const float hn = sigf(ov) * tanhf(cn);
        cbuf[cidx] = cn;
        hdst[(size_t)m * H_DIM + j] = (__bf16)hn;
      }
    }
  } else {
    // ---------------- logits "layer" (l = 12) ----------------
    const int t = d - L_LAYERS;
    if (t < 0 || t >= T_STEPS) return;
    const int wg = blockIdx.x - 96;
    const int nt = wg * 8 + wave;
    if (nt >= NT_OUT) return;

    const __bf16* hrows =
        hbuf + parR * HSLAB + (size_t)(L_LAYERS - 1) * B_BATCH * H_DIM;
    const __bf16* wb = WoutSw + (size_t)nt * 16 * 512;

    v8f acc[4];
    const v8f zero = {0.f, 0.f, 0.f, 0.f, 0.f, 0.f, 0.f, 0.f};
    for (int q = 0; q < 4; ++q) acc[q] = zero;

    for (int kc = 0; kc < 16; ++kc) {
      const int kb = kc * 32;
      const v16bf bf = *(const v16bf*)(wb + (size_t)kc * 512 + lane * 16);
      for (int mt = 0; mt < 4; ++mt) {
        v16bf a = load_afrag(hrows + (size_t)(mt * 16 + lm) * 512, kb, g);
        acc[mt] = wmma_bf16(a, bf, acc[mt]);
      }
    }

    const int v = nt * 16 + lm;
    if (v < V_VOCAB) {
      const float bo = b_out[v];
      for (int mt = 0; mt < 4; ++mt) {
        for (int r = 0; r < 8; ++r) {
          const int m = mt * 16 + ((lane < 16) ? r : (r + 8));
          out[(size_t)m * T_STEPS * V_VOCAB + (size_t)t * V_VOCAB + v] =
              acc[mt][r] + bo;
        }
      }
    }
  }
}

// ---------------------------------------------------------------------------
extern "C" void kernel_launch(void* const* d_in, const int* in_sizes, int n_in,
                              void* d_out, int out_size, void* d_ws,
                              size_t ws_size, hipStream_t stream) {
  const float* h0 = (const float*)d_in[0];
  const float* c0 = (const float*)d_in[1];
  const int* tok = (const int*)d_in[2];
  const float* table = (const float*)d_in[3];
  const float* W_ih = (const float*)d_in[4];
  const float* W_hh = (const float*)d_in[5];
  const float* b_ih = (const float*)d_in[6];
  const float* b_hh = (const float*)d_in[7];
  const float* W_out = (const float*)d_in[8];
  const float* b_out = (const float*)d_in[9];
  float* out = (float*)d_out;

  char* ws = (char*)d_ws;
  __bf16* Wsw = (__bf16*)(ws + WS_WSW);
  __bf16* WoutSw = (__bf16*)(ws + WS_WOUT);
  __bf16* Xemb = (__bf16*)(ws + WS_XEMB);
  __bf16* hbuf = (__bf16*)(ws + WS_HBUF);
  float* cbuf = (float*)(ws + WS_CBUF);

  // Setup: weight swizzle (bf16 fragment order), embedding gather, state init.
  swizzle_weights<<<L_LAYERS * 32 * 4 * 32 + NT_OUT * 16, 256, 0, stream>>>(
      W_ih, W_hh, W_out, Wsw, WoutSw);
  embed_tokens<<<T_STEPS * B_BATCH, 256, 0, stream>>>(tok, table, Xemb);
  init_state<<<(L_LAYERS * B_BATCH * H_DIM + 255) / 256, 256, 0, stream>>>(
      h0, c0, hbuf, cbuf);

  // Wavefront over diagonals d = l + t; logits ride as layer 12.
  for (int d = 0; d < T_STEPS + L_LAYERS; ++d) {
    diag_step<<<104, 256, 0, stream>>>(d, b_ih, b_hh, b_out, Wsw, WoutSw, Xemb,
                                       hbuf, cbuf, out);
  }
}
